// Transformer_Block_25615184954012
// MI455X (gfx1250) — matrix operand order repaired and checked by reading, not compiled, not run on hardware
//
#include <hip/hip_runtime.h>
#include <hip/hip_bf16.h>
#include <cstdint>

#define B_   2
#define T_   2048
#define D_   1024
#define H_   16
#define DH_  64
#define DFF_ 4096
#define BT_  (B_ * T_)

typedef __bf16 bf16;
typedef __attribute__((ext_vector_type(16))) __bf16 v16bf;
typedef __attribute__((ext_vector_type(8)))  float  v8f;

union Frag {
  v16bf v;
  uint4 q[2];
  bf16  h[16];
};

// ---------------------------------------------------------------------------
// CDNA5 async global->LDS copy (ASYNCcnt tracked) + wait.
// ---------------------------------------------------------------------------
__device__ __forceinline__ void async_ld_b128(unsigned lds_off, const void* gptr) {
  asm volatile("global_load_async_to_lds_b128 %0, %1, off"
               :: "v"(lds_off), "v"(gptr)
               : "memory");
}
__device__ __forceinline__ void wait_async0() {
  asm volatile("s_wait_asynccnt 0x0" ::: "memory");
}
__device__ __forceinline__ unsigned lds_addr(const void* p) {
  return (unsigned)(uintptr_t)p;   // low 32 bits of generic LDS addr = LDS offset
}

// ---------------------------------------------------------------------------
// Tiled transpose+convert: src[K][N] fp32 -> dst[N][K] bf16.
// 32x32 tiles via padded LDS: both global streams fully coalesced.
// ---------------------------------------------------------------------------
__global__ __launch_bounds__(256) void transpose_tile(const float* __restrict__ src,
                                                      bf16* __restrict__ dst,
                                                      int K, int N) {
  __shared__ float tile[32][33];
  const int n0 = blockIdx.x * 32, k0 = blockIdx.y * 32;
  const int tx = threadIdx.x, ty = threadIdx.y;   // 32 x 8
#pragma unroll
  for (int i = 0; i < 32; i += 8)
    tile[ty + i][tx] = src[(size_t)(k0 + ty + i) * N + n0 + tx];
  __syncthreads();
#pragma unroll
  for (int i = 0; i < 32; i += 8)
    dst[(size_t)(n0 + ty + i) * K + k0 + tx] = (bf16)tile[tx][ty + i];
}

// Per-head QKV weight pack: each (which, head) is a [D][DH] fp32 slab
// transposed into rows (which*D + h*DH .. +DH) of Wt[3D][D] bf16.
__global__ __launch_bounds__(256) void pack_qkv_w_tiled(const float* __restrict__ Wq,
                                                        const float* __restrict__ Wk,
                                                        const float* __restrict__ Wv,
                                                        bf16* __restrict__ Wt) {
  __shared__ float tile[32][33];
  const int slab  = blockIdx.z;           // 0..47
  const int which = slab >> 4, hh = slab & 15;
  const float* src = ((which == 0) ? Wq : (which == 1) ? Wk : Wv) +
                     (size_t)hh * D_ * DH_;                    // [D][64]
  bf16* dst = Wt + (size_t)(which * D_ + hh * DH_) * D_;       // [64][D]
  const int n0 = blockIdx.x * 32, k0 = blockIdx.y * 32;        // n<64, k<D
  const int tx = threadIdx.x, ty = threadIdx.y;
#pragma unroll
  for (int i = 0; i < 32; i += 8)
    tile[ty + i][tx] = src[(size_t)(k0 + ty + i) * DH_ + n0 + tx];
  __syncthreads();
#pragma unroll
  for (int i = 0; i < 32; i += 8)
    dst[(size_t)(n0 + ty + i) * D_ + k0 + tx] = (bf16)tile[tx][ty + i];
}

__global__ void pack_bias(const float* __restrict__ bq,
                          const float* __restrict__ bk,
                          const float* __restrict__ bv,
                          float* __restrict__ bqkv) {
  int i = blockIdx.x * 256 + threadIdx.x;
  if (i >= 3 * D_) return;
  bqkv[i] = (i < D_) ? bq[i] : (i < 2 * D_) ? bk[i - D_] : bv[i - 2 * D_];
}

// ---------------------------------------------------------------------------
// LayerNorm: one row (D=1024) per 256-thread block, wave32 shuffle reductions.
// ---------------------------------------------------------------------------
__global__ __launch_bounds__(256) void ln_kernel(const float* __restrict__ x,
                                                 const float* __restrict__ g,
                                                 const float* __restrict__ b,
                                                 bf16* __restrict__ out) {
  __shared__ float red[8];
  const int row = blockIdx.x;
  const int t   = threadIdx.x;
  const float* xr = x + (size_t)row * D_;
  float v[4];
  float s = 0.f;
#pragma unroll
  for (int i = 0; i < 4; ++i) { v[i] = xr[t + 256 * i]; s += v[i]; }
#pragma unroll
  for (int d = 16; d >= 1; d >>= 1) s += __shfl_xor(s, d, 32);
  if ((t & 31) == 0) red[t >> 5] = s;
  __syncthreads();
  float tot = 0.f;
#pragma unroll
  for (int i = 0; i < 8; ++i) tot += red[i];
  const float mu = tot * (1.f / D_);
  __syncthreads();
  float vs = 0.f;
#pragma unroll
  for (int i = 0; i < 4; ++i) { float d0 = v[i] - mu; vs += d0 * d0; }
#pragma unroll
  for (int d = 16; d >= 1; d >>= 1) vs += __shfl_xor(vs, d, 32);
  if ((t & 31) == 0) red[t >> 5] = vs;
  __syncthreads();
  float vtot = 0.f;
#pragma unroll
  for (int i = 0; i < 8; ++i) vtot += red[i];
  const float inv = rsqrtf(vtot * (1.f / D_) + 1e-5f);
  bf16* orow = out + (size_t)row * D_;
#pragma unroll
  for (int i = 0; i < 4; ++i) {
    int c = t + 256 * i;
    orow[c] = (bf16)((v[i] - mu) * inv * g[c] + b[c]);
  }
}

// ---------------------------------------------------------------------------
// Tiled bf16 GEMM: C[M,N] = A[M,K] * Bt[N,K]^T (+bias, +resid / relu).
// Block tile 128x256x32, 8 waves (2x4), wave tile 64x64 -> 16 WMMA / K-step.
// Double-buffered LDS fed by global_load_async_to_lds_b128.
// MODE 0: fp32 out + residual     (Wo, FFN2)
// MODE 1: bf16 out + ReLU         (FFN1)
// MODE 2: bf16 scatter: Q(*0.125)/K -> [B,H,T,DH], V -> [B,H,DH,T]
// ---------------------------------------------------------------------------
template <int MODE>
__global__ __launch_bounds__(256) void gemm_bf16_kernel(
    const bf16* __restrict__ A, const bf16* __restrict__ Bt,
    const float* __restrict__ bias, const float* resid,
    float* outf, bf16* __restrict__ outb, int M, int N, int K) {
  __shared__ bf16 As[2][128][40];   // pad 32 -> 40 (20 banks): conflict-free
  __shared__ bf16 Bs[2][256][40];
  const int t     = threadIdx.x;
  const int lane  = t & 31;
  const int wave  = t >> 5;
  const int half  = lane >> 4;
  const int l16   = lane & 15;
  const int waveM = wave >> 2;          // 0..1  (64-row strip)
  const int waveN = wave & 3;           // 0..3  (64-col strip)
  const int m0 = blockIdx.y * 128;
  const int n0 = blockIdx.x * 256;
  const int lrA = t >> 1;               // 0..127
  const int lcA = (t & 1) * 16;         // 0 or 16
  const int lrB = t;                    // 0..255

  const bf16* gA = A  + (size_t)(m0 + lrA) * K + lcA;
  const bf16* gB = Bt + (size_t)(n0 + lrB) * K;
  unsigned la[2] = {lds_addr(&As[0][lrA][lcA]), lds_addr(&As[1][lrA][lcA])};
  unsigned lb[2] = {lds_addr(&Bs[0][lrB][0]),   lds_addr(&Bs[1][lrB][0])};

  auto issue = [&](int buf, int k0) {
    async_ld_b128(la[buf],      gA + k0);
    async_ld_b128(la[buf] + 16, gA + k0 + 8);
    async_ld_b128(lb[buf],      gB + k0);
    async_ld_b128(lb[buf] + 16, gB + k0 + 8);
    async_ld_b128(lb[buf] + 32, gB + k0 + 16);
    async_ld_b128(lb[buf] + 48, gB + k0 + 24);
  };

  v8f acc[4][4];
#pragma unroll
  for (int mi = 0; mi < 4; ++mi)
#pragma unroll
    for (int ni = 0; ni < 4; ++ni) acc[mi][ni] = 0.f;

  issue(0, 0);
  int cur = 0;
  for (int k0 = 0; k0 < K; k0 += 32) {
    wait_async0();
    __syncthreads();
    if (k0 + 32 < K) issue(cur ^ 1, k0 + 32);

    Frag af[4], bfr[4];
#pragma unroll
    for (int mi = 0; mi < 4; ++mi) {
      const bf16* p = &As[cur][waveM * 64 + mi * 16 + l16][0];
      af[mi].q[0] = *(const uint4*)(p + 16 * half);        // K = 16h + e
      af[mi].q[1] = *(const uint4*)(p + 16 * half + 8);    // K = 16h + 8 + e
    }
#pragma unroll
    for (int ni = 0; ni < 4; ++ni) {
      const bf16* p = &Bs[cur][waveN * 64 + ni * 16 + l16][16 * half];  // K=16h+e
      bfr[ni].q[0] = ((const uint4*)p)[0];
      bfr[ni].q[1] = ((const uint4*)p)[1];
    }
#pragma unroll
    for (int mi = 0; mi < 4; ++mi)
#pragma unroll
      for (int ni = 0; ni < 4; ++ni)
        acc[mi][ni] = __builtin_amdgcn_wmma_f32_16x16x32_bf16(
            false, af[mi].v, false, bfr[ni].v, (short)0, acc[mi][ni], false, false);
    cur ^= 1;
  }

#pragma unroll
  for (int mi = 0; mi < 4; ++mi) {
#pragma unroll
    for (int ni = 0; ni < 4; ++ni) {
#pragma unroll
      for (int r = 0; r < 8; ++r) {
        const int gm = m0 + waveM * 64 + mi * 16 + 8 * half + r;
        const int gn = n0 + waveN * 64 + ni * 16 + l16;
        float v = acc[mi][ni][r] + bias[gn];
        if (MODE == 1) {
          v = fmaxf(v, 0.f);
          outb[(size_t)gm * N + gn] = (bf16)v;
        } else if (MODE == 0) {
          outf[(size_t)gm * N + gn] = v + resid[(size_t)gm * N + gn];
        } else {  // MODE == 2 : fused QKV scatter
          const int which = gn >> 10;
          const int hn = gn & (D_ - 1);
          const int hh = hn >> 6, e = hn & (DH_ - 1);
          const int bb = gm >> 11, tt = gm & (T_ - 1);
          size_t idx;
          if (which == 2) {      // V stored transposed [B,H,DH,T]
            idx = 2 * ((size_t)B_ * H_ * T_ * DH_) +
                  (((size_t)bb * H_ + hh) * DH_ + e) * T_ + tt;
          } else {               // Q (pre-scaled by 1/sqrt(DH)) and K: [B,H,T,DH]
            if (which == 0) v *= 0.125f;
            idx = (size_t)which * ((size_t)B_ * H_ * T_ * DH_) +
                  (((size_t)bb * H_ + hh) * T_ + tt) * DH_ + e;
          }
          outb[idx] = (bf16)v;
        }
      }
    }
  }
}

// ---------------------------------------------------------------------------
// Flash attention: block = (64 q-rows, 1 head, 1 batch), 4 waves x 16 rows.
// K [key][dh] and V^T [dh][key] tiles double-buffered in LDS via async loads.
// S = Q*K^T and O += P*V via v_wmma_f32_16x16x32_bf16.
// ---------------------------------------------------------------------------
__global__ __launch_bounds__(128, 2) void attn_kernel(
    const bf16* __restrict__ Qg, const bf16* __restrict__ Kg,
    const bf16* __restrict__ Vtg, bf16* __restrict__ Aout) {
  __shared__ bf16 Ks[2][64][72];
  __shared__ bf16 Vs[2][64][72];      // V^T tile: [dh][key]
  __shared__ bf16 Ps[4][16][72];      // per-wave P strip [16 rows][64 keys]
  const int qt = blockIdx.x, h = blockIdx.y, b = blockIdx.z;
  const int lane = threadIdx.x & 31, wave = threadIdx.x >> 5;
  const int half = lane >> 4, l16 = lane & 15;
  const size_t base = (((size_t)b * H_) + h) * T_ * DH_;
  const bf16* Qp = Qg + base;
  const int q0 = qt * 64;
  const int qrowA = q0 + wave * 16 + l16;

  // Q A-fragments (pre-scaled by 1/sqrt(DH) at QKV epilogue).
  Frag qa[2];
#pragma unroll
  for (int ks = 0; ks < 2; ++ks) {
    const bf16* p = Qp + (size_t)qrowA * DH_ + ks * 32;
    qa[ks].q[0] = *(const uint4*)(p + 16 * half);
    qa[ks].q[1] = *(const uint4*)(p + 16 * half + 8);
  }

  v8f o[4];
#pragma unroll
  for (int nt = 0; nt < 4; ++nt) o[nt] = 0.f;
  float mr[8], lsum[8];
#pragma unroll
  for (int r = 0; r < 8; ++r) { mr[r] = -1e30f; lsum[r] = 0.f; }

  const int ldr = threadIdx.x >> 1;        // 0..63
  const int ldc = (threadIdx.x & 1) * 32;  // 0 or 32
  const bf16* Kbase = Kg + base + (size_t)ldr * DH_ + ldc;   // + kt*64*DH_
  const bf16* Vbase = Vtg + base + (size_t)ldr * T_ + ldc;   // + kt*64
  unsigned lk[2] = {lds_addr(&Ks[0][ldr][ldc]), lds_addr(&Ks[1][ldr][ldc])};
  unsigned lv[2] = {lds_addr(&Vs[0][ldr][ldc]), lds_addr(&Vs[1][ldr][ldc])};

  auto issue = [&](int buf, int kt) {
#pragma unroll
    for (int i = 0; i < 4; ++i) {
      async_ld_b128(lk[buf] + 16 * i, Kbase + (size_t)kt * 64 * DH_ + 8 * i);
      async_ld_b128(lv[buf] + 16 * i, Vbase + (size_t)kt * 64 + 8 * i);
    }
  };

  issue(0, 0);
  int cur = 0;
  for (int kt = 0; kt <= qt; ++kt) {
    wait_async0();
    __syncthreads();
    if (kt < qt) issue(cur ^ 1, kt + 1);

    // S strip [16 x 64]
    v8f s[4];
#pragma unroll
    for (int nt = 0; nt < 4; ++nt) {
      v8f acc = 0.f;
#pragma unroll
      for (int ks = 0; ks < 2; ++ks) {
        Frag kb;  // B[k][n] = K[key=n][dh=k] -> contiguous along dh
        const bf16* p = &Ks[cur][nt * 16 + l16][ks * 32 + 16 * half];
        kb.q[0] = ((const uint4*)p)[0];
        kb.q[1] = ((const uint4*)p)[1];
        acc = __builtin_amdgcn_wmma_f32_16x16x32_bf16(
            false, qa[ks].v, false, kb.v, (short)0, acc, false, false);
      }
      s[nt] = acc;
    }

    // causal mask + online softmax (row = 8*half + r, cols on lanes)
#pragma unroll
    for (int r = 0; r < 8; ++r) {
      const int qi = q0 + wave * 16 + 8 * half + r;
      float mx = mr[r];
#pragma unroll
      for (int nt = 0; nt < 4; ++nt) {
        const int kj = kt * 64 + nt * 16 + l16;
        float val = s[nt][r];
        if (kj > qi) val = -1e30f;
        s[nt][r] = val;
        mx = fmaxf(mx, val);
      }
#pragma unroll
      for (int d = 1; d < 16; d <<= 1) mx = fmaxf(mx, __shfl_xor(mx, d, 32));
      const float corr = __expf(mr[r] - mx);
      float rs = 0.f;
#pragma unroll
      for (int nt = 0; nt < 4; ++nt) {
        float e = __expf(s[nt][r] - mx);
        s[nt][r] = e;
        rs += e;
      }
#pragma unroll
      for (int d = 1; d < 16; d <<= 1) rs += __shfl_xor(rs, d, 32);
      lsum[r] = lsum[r] * corr + rs;
      mr[r] = mx;
#pragma unroll
      for (int nt = 0; nt < 4; ++nt) o[nt][r] *= corr;
      // C-layout -> LDS (wave-private strip; same-wave LDS ops are in-order)
#pragma unroll
      for (int nt = 0; nt < 4; ++nt)
        Ps[wave][8 * half + r][nt * 16 + l16] = (bf16)s[nt][r];
    }

    // O += P * V   (V fragment contiguous thanks to V^T layout)
#pragma unroll
    for (int ks = 0; ks < 2; ++ks) {
      Frag pa;
      const bf16* p = &Ps[wave][l16][ks * 32];
      pa.q[0] = *(const uint4*)(p + 16 * half);
      pa.q[1] = *(const uint4*)(p + 16 * half + 8);
#pragma unroll
      for (int nt = 0; nt < 4; ++nt) {
        Frag vb;  // B[k=key][n=dh] = Vs[dh][key]
        const bf16* pv = &Vs[cur][nt * 16 + l16][ks * 32 + 16 * half];
        vb.q[0] = ((const uint4*)pv)[0];
        vb.q[1] = ((const uint4*)pv)[1];
        o[nt] = __builtin_amdgcn_wmma_f32_16x16x32_bf16(
            false, pa.v, false, vb.v, (short)0, o[nt], false, false);
      }
    }
    cur ^= 1;
  }

  // write O / l  -> concat-head layout [B,T,D] bf16
#pragma unroll
  for (int r = 0; r < 8; ++r) {
    const int qi = q0 + wave * 16 + 8 * half + r;
    const float invl = 1.f / lsum[r];
#pragma unroll
    for (int nt = 0; nt < 4; ++nt) {
      const int col = h * DH_ + nt * 16 + l16;
      Aout[((size_t)b * T_ + qi) * D_ + col] = (bf16)(o[nt][r] * invl);
    }
  }
}

// ---------------------------------------------------------------------------
// Launch
// ---------------------------------------------------------------------------
extern "C" void kernel_launch(void* const* d_in, const int* in_sizes, int n_in,
                              void* d_out, int out_size, void* d_ws,
                              size_t ws_size, hipStream_t stream) {
  const float* x    = (const float*)d_in[0];
  const float* Wq   = (const float*)d_in[1];
  const float* bq   = (const float*)d_in[2];
  const float* Wk   = (const float*)d_in[3];
  const float* bk   = (const float*)d_in[4];
  const float* Wv   = (const float*)d_in[5];
  const float* bv   = (const float*)d_in[6];
  const float* Wo   = (const float*)d_in[7];
  const float* bo   = (const float*)d_in[8];
  const float* ln1g = (const float*)d_in[9];
  const float* ln1b = (const float*)d_in[10];
  const float* ln2g = (const float*)d_in[11];
  const float* ln2b = (const float*)d_in[12];
  const float* W1   = (const float*)d_in[13];
  const float* b1   = (const float*)d_in[14];
  const float* W2   = (const float*)d_in[15];
  const float* b2   = (const float*)d_in[16];
  float* out = (float*)d_out;

  char* ws = (char*)d_ws;
  size_t off = 0;
  auto alloc = [&](size_t bytes) -> void* {
    void* p = ws + off;
    off += (bytes + 255) & ~(size_t)255;
    return p;
  };
  bf16*  h1     = (bf16*)alloc((size_t)BT_ * D_ * 2);
  bf16*  h2     = (bf16*)alloc((size_t)BT_ * D_ * 2);
  bf16*  Wqkv_t = (bf16*)alloc((size_t)3 * D_ * D_ * 2);
  float* bqkv   = (float*)alloc((size_t)3 * D_ * 4);
  bf16*  Wo_t   = (bf16*)alloc((size_t)D_ * D_ * 2);
  bf16*  W1_t   = (bf16*)alloc((size_t)D_ * DFF_ * 2);
  bf16*  W2_t   = (bf16*)alloc((size_t)DFF_ * D_ * 2);
  bf16*  qkv    = (bf16*)alloc((size_t)3 * B_ * H_ * T_ * DH_ * 2);
  bf16*  attn_a = (bf16*)alloc((size_t)BT_ * D_ * 2);
  bf16*  ffh    = (bf16*)alloc((size_t)BT_ * DFF_ * 2);
  (void)in_sizes; (void)n_in; (void)out_size; (void)ws_size;

  // weight packing (bf16, transposed, all streams coalesced via LDS tiles)
  dim3 tb(32, 8);
  pack_qkv_w_tiled<<<dim3(DH_ / 32, D_ / 32, 48), tb, 0, stream>>>(Wq, Wk, Wv, Wqkv_t);
  pack_bias<<<(3 * D_ + 255) / 256, 256, 0, stream>>>(bq, bk, bv, bqkv);
  transpose_tile<<<dim3(D_ / 32, D_ / 32), tb, 0, stream>>>(Wo, Wo_t, D_, D_);
  transpose_tile<<<dim3(DFF_ / 32, D_ / 32), tb, 0, stream>>>(W1, W1_t, D_, DFF_);
  transpose_tile<<<dim3(D_ / 32, DFF_ / 32), tb, 0, stream>>>(W2, W2_t, DFF_, D_);

  // LN1 -> h1 (bf16)
  ln_kernel<<<BT_, 256, 0, stream>>>(x, ln1g, ln1b, h1);

  // fused QKV projection -> qkv (Q,K: [B,H,T,DH], V: [B,H,DH,T]) bf16
  gemm_bf16_kernel<2><<<dim3(3 * D_ / 256, BT_ / 128), 256, 0, stream>>>(
      h1, Wqkv_t, bqkv, nullptr, nullptr, qkv, BT_, 3 * D_, D_);

  // flash attention -> attn_a [BT, D] bf16 (heads concatenated)
  const size_t hsz = (size_t)B_ * H_ * T_ * DH_;
  attn_kernel<<<dim3(T_ / 64, H_, B_), 128, 0, stream>>>(
      qkv, qkv + hsz, qkv + 2 * hsz, attn_a);

  // Wo projection + bo + residual x -> out (fp32)
  gemm_bf16_kernel<0><<<dim3(D_ / 256, BT_ / 128), 256, 0, stream>>>(
      attn_a, Wo_t, bo, x, out, nullptr, BT_, D_, D_);

  // LN2 -> h2 (bf16)
  ln_kernel<<<BT_, 256, 0, stream>>>(out, ln2g, ln2b, h2);

  // FFN1 + b1 + ReLU -> ffh (bf16)
  gemm_bf16_kernel<1><<<dim3(DFF_ / 256, BT_ / 128), 256, 0, stream>>>(
      h2, W1_t, b1, nullptr, nullptr, ffh, BT_, DFF_, D_);

  // FFN2 + b2 + residual(out) -> out (fp32)
  gemm_bf16_kernel<0><<<dim3(D_ / 256, BT_ / 128), 256, 0, stream>>>(
      ffh, W2_t, b2, out, out, nullptr, BT_, D_, DFF_);
}
